// MultiheadAttention_28295244546681
// MI455X (gfx1250) — compile-verified, matrix operand order checked by reading, not run on hardware
//
#include <hip/hip_runtime.h>
#include <stdint.h>

typedef __bf16 bf16;
typedef __attribute__((ext_vector_type(16))) __bf16 v16bf;
typedef __attribute__((ext_vector_type(4)))  __bf16 v4bf;
typedef __attribute__((ext_vector_type(8)))  float  v8f;
typedef __attribute__((ext_vector_type(4)))  int    i4vec;

#define B_  2
#define S_  2048
#define D_  1024
#define H_  16
#define HD_ 64
#define M_  (B_*S_)   /* 4096 rows of the "token" matrices */

// ---------------------------------------------------------------------------
// Async global->LDS staging (CDNA5 GLOBAL_LOAD_ASYNC_TO_LDS_B128, ASYNCcnt),
// with a synchronous fallback so compilation never regresses.
// Builtin signature (from clang diagnostic): (int4 AS1*, int4 AS3*, Ii, Ii).
// ---------------------------------------------------------------------------
#if __has_builtin(__builtin_amdgcn_global_load_async_to_lds_b128)
#define HAVE_ASYNC 1
#else
#define HAVE_ASYNC 0
#endif

__device__ __forceinline__ void copy16(const bf16* __restrict__ g,
                                       bf16* __restrict__ l) {
#if HAVE_ASYNC
  __builtin_amdgcn_global_load_async_to_lds_b128(
      (__attribute__((address_space(1))) i4vec*)g,
      (__attribute__((address_space(3))) i4vec*)l, 0, 0);
#else
  *(uint4*)l = *(const uint4*)g;
#endif
}

__device__ __forceinline__ void wait_async0() {
#if HAVE_ASYNC
#if __has_builtin(__builtin_amdgcn_s_wait_asynccnt)
  __builtin_amdgcn_s_wait_asynccnt(0);
#else
  asm volatile("s_wait_asynccnt 0x0" ::: "memory");
#endif
#endif
}

// ---------------------------------------------------------------------------
// Fragment helpers (wave32 WMMA, 16x16x32 bf16)
// A-fragment of a row-major MxK tile: lane L holds row (L&15); its 16 bf16
// are K = {0..7,16..23} (lanes 0-15) or {8..15,24..31} (lanes 16-31), i.e.
// two contiguous 16-byte loads.  Same gather on row-major B^T (NxK) gives
// the B-fragment.
// ---------------------------------------------------------------------------
union FragU { v16bf v; uint4 q[2]; };

__device__ __forceinline__ v16bf load_frag(const bf16* __restrict__ base, int ld) {
  int lane = threadIdx.x & 31;
  int r  = lane & 15;
  int kb = (lane >> 4) << 3;          // 0 or 8
  FragU f;
  f.q[0] = *(const uint4*)(base + r * ld + kb);
  f.q[1] = *(const uint4*)(base + r * ld + kb + 16);
  return f.v;
}

__device__ __forceinline__ v8f wmma_bf16(v16bf a, v16bf b, v8f c) {
  return __builtin_amdgcn_wmma_f32_16x16x32_bf16(false, a, false, b,
                                                 (short)0, c, false, false);
}

// ---------------------------------------------------------------------------
// DPP16 reductions across each 16-lane half-wave (no LDS traffic):
// quad_perm(1,0,3,2)=0xB1, quad_perm(2,3,0,1)=0x4E,
// row_half_mirror=0x141, row_mirror=0x140.
// ---------------------------------------------------------------------------
template<int CTRL>
__device__ __forceinline__ float dpp_mov(float x) {
  return __builtin_bit_cast(float,
      __builtin_amdgcn_update_dpp(0, __builtin_bit_cast(int, x),
                                  CTRL, 0xF, 0xF, true));
}
__device__ __forceinline__ float red_max16(float v) {
  v = fmaxf(v, dpp_mov<0xB1>(v));
  v = fmaxf(v, dpp_mov<0x4E>(v));
  v = fmaxf(v, dpp_mov<0x141>(v));
  v = fmaxf(v, dpp_mov<0x140>(v));
  return v;
}
__device__ __forceinline__ float red_sum16(float v) {
  v += dpp_mov<0xB1>(v);
  v += dpp_mov<0x4E>(v);
  v += dpp_mov<0x141>(v);
  v += dpp_mov<0x140>(v);
  return v;
}

// ---------------------------------------------------------------------------
// f32 -> bf16 conversion (bandwidth-bound one-shot pass)
// ---------------------------------------------------------------------------
__global__ __launch_bounds__(256) void cvt_f32_bf16(const float4* __restrict__ src,
                                                    v4bf* __restrict__ dst, int n4) {
  int i = blockIdx.x * 256 + threadIdx.x;
  if (i < n4) {
    float4 f = src[i];
    v4bf o = { (bf16)f.x, (bf16)f.y, (bf16)f.z, (bf16)f.w };
    dst[i] = o;
  }
}

// ---------------------------------------------------------------------------
// GEMM: C[M,1024] = A_bf16[M,1024] @ W_bf16[N=1024,K=1024]^T + bias
// WG tile 128(M) x 64(N), 8 waves; K staged 64 at a time, double-buffered
// through LDS with async copies.  8 WMMAs per wave per stage, 16 stages.
// EPI: 0 = Q (scale 1/8 -> [B,H,S,HD] bf16)   1 = K (-> [B,H,S,HD] bf16)
//      2 = V (-> transposed [B,H,HD,S] bf16)  3 = O (-> f32 d_out)
// ---------------------------------------------------------------------------
template<int EPI>
__global__ __launch_bounds__(256) void gemm_bf16_kernel(
    const bf16* __restrict__ A, const bf16* __restrict__ Bw,
    const float* __restrict__ bias, bf16* __restrict__ outb,
    float* __restrict__ outf) {
  const int K = D_;
  __shared__ __align__(16) bf16 As[2][128 * 64];   // 2 x 16 KB
  __shared__ __align__(16) bf16 Bs[2][64 * 64];    // 2 x  8 KB

  const int tid  = threadIdx.x;
  const int wave = tid >> 5;
  const int lane = tid & 31;
  const int m0 = blockIdx.y * 128;
  const int n0 = blockIdx.x * 64;

  auto stage = [&](int buf, int kk) {
#pragma unroll
    for (int i = 0; i < 4; ++i) {                  // A: 128x64 = 1024 uint4
      int id = tid + i * 256;
      int row = id >> 3, c8 = id & 7;
      copy16(A + (size_t)(m0 + row) * K + kk + c8 * 8,
             &As[buf][row * 64 + c8 * 8]);
    }
#pragma unroll
    for (int i = 0; i < 2; ++i) {                  // B^T: 64x64 = 512 uint4
      int id = tid + i * 256;
      int row = id >> 3, c8 = id & 7;
      copy16(Bw + (size_t)(n0 + row) * K + kk + c8 * 8,
             &Bs[buf][row * 64 + c8 * 8]);
    }
  };

  v8f acc[4] = { {0,0,0,0,0,0,0,0}, {0,0,0,0,0,0,0,0},
                 {0,0,0,0,0,0,0,0}, {0,0,0,0,0,0,0,0} };

  stage(0, 0);
  const int NIT = K / 64;                           // 16 stages
  for (int it = 0; it < NIT; ++it) {
    const int cur = it & 1;
    wait_async0();
    __syncthreads();                                // stage ready + prev compute done
    if (it + 1 < NIT) stage(cur ^ 1, (it + 1) * 64);

    v16bf a0 = load_frag(&As[cur][(wave * 16) * 64],      64);
    v16bf a1 = load_frag(&As[cur][(wave * 16) * 64 + 32], 64);
#pragma unroll
    for (int j = 0; j < 4; ++j) {
      v16bf b0 = load_frag(&Bs[cur][(j * 16) * 64],      64);
      v16bf b1 = load_frag(&Bs[cur][(j * 16) * 64 + 32], 64);
      acc[j] = wmma_bf16(a0, b0, acc[j]);
      acc[j] = wmma_bf16(a1, b1, acc[j]);
    }
  }

  // epilogue: C-fragment element (j, r) -> row (r + 8*half), col (lane&15)
  const int nl = lane & 15;
  const int mh = (lane >> 4) * 8;
#pragma unroll
  for (int j = 0; j < 4; ++j) {
#pragma unroll
    for (int r = 0; r < 8; ++r) {
      int m = m0 + wave * 16 + r + mh;              // token row: b*S + s
      int n = n0 + j * 16 + nl;                     // feature:   h*HD + d
      float v = acc[j][r] + bias[n];
      int b = m >> 11, s = m & (S_ - 1);
      int h = n >> 6,  d = n & (HD_ - 1);
      if (EPI == 0) {
        outb[(((size_t)(b * H_ + h) * S_ + s) * HD_) + d] = (bf16)(v * 0.125f);
      } else if (EPI == 1) {
        outb[(((size_t)(b * H_ + h) * S_ + s) * HD_) + d] = (bf16)v;
      } else if (EPI == 2) {
        outb[(((size_t)(b * H_ + h) * HD_ + d) * S_) + s] = (bf16)v;
      } else {
        outf[(size_t)m * D_ + n] = v;
      }
    }
  }
}

// ---------------------------------------------------------------------------
// Flash attention: WG = one (b,h) x 128 query rows; 8 waves x 16 rows.
// KV chunks of 64, double-buffered async staging, one barrier per chunk,
// DPP softmax reductions.  Scale pre-folded into Q.
// ---------------------------------------------------------------------------
__global__ __launch_bounds__(256) void attn_kernel(
    const bf16* __restrict__ qh, const bf16* __restrict__ kh,
    const bf16* __restrict__ vT, bf16* __restrict__ ctx) {
  __shared__ __align__(16) bf16 Kt[2][64 * 64];     // [key][d]   2 x 8 KB
  __shared__ __align__(16) bf16 Vt[2][64 * 64];     // [d][key]   2 x 8 KB
  __shared__ __align__(16) bf16 Pw[8 * 16 * 64];    // per-wave P scratch 16 KB

  const int tid  = threadIdx.x;
  const int wave = tid >> 5;
  const int lane = tid & 31;
  const int bh = blockIdx.y;                        // b*H + h
  const int b  = bh >> 4, h = bh & (H_ - 1);
  const int qrow0 = blockIdx.x * 128 + wave * 16;

  const bf16* qbase = qh + (size_t)bh * S_ * HD_;
  const bf16* kbase = kh + (size_t)bh * S_ * HD_;
  const bf16* vbase = vT + (size_t)bh * HD_ * S_;

  auto stage = [&](int buf, int s0) {
#pragma unroll
    for (int i = 0; i < 2; ++i) {
      int id = tid + i * 256;
      int row = id >> 3, c8 = id & 7;
      copy16(kbase + (size_t)(s0 + row) * HD_ + c8 * 8,
             &Kt[buf][row * 64 + c8 * 8]);
      copy16(vbase + (size_t)row * S_ + s0 + c8 * 8,
             &Vt[buf][row * 64 + c8 * 8]);
    }
  };

  // Q A-fragments live in registers for the whole KV sweep
  v16bf qa0 = load_frag(qbase + (size_t)qrow0 * HD_,      HD_);
  v16bf qa1 = load_frag(qbase + (size_t)qrow0 * HD_ + 32, HD_);

  v8f acc_o[4] = { {0,0,0,0,0,0,0,0}, {0,0,0,0,0,0,0,0},
                   {0,0,0,0,0,0,0,0}, {0,0,0,0,0,0,0,0} };
  float m_i[8], l_i[8];
#pragma unroll
  for (int r = 0; r < 8; ++r) { m_i[r] = -3.0e38f; l_i[r] = 0.0f; }

  const int nl = lane & 15;
  const int mh = (lane >> 4) * 8;
  bf16* pw = Pw + wave * 16 * 64;

  stage(0, 0);
  const int NC = S_ / 64;                            // 32 chunks
  for (int c = 0; c < NC; ++c) {
    const int cur = c & 1;
    wait_async0();
    __syncthreads();                                 // data ready + prev compute done
    if (c + 1 < NC) stage(cur ^ 1, (c + 1) * 64);

    // S = Q K^T
    v8f sacc[4] = { {0,0,0,0,0,0,0,0}, {0,0,0,0,0,0,0,0},
                    {0,0,0,0,0,0,0,0}, {0,0,0,0,0,0,0,0} };
#pragma unroll
    for (int j = 0; j < 4; ++j) {
      v16bf kb0 = load_frag(&Kt[cur][(j * 16) * 64],      64);
      v16bf kb1 = load_frag(&Kt[cur][(j * 16) * 64 + 32], 64);
      sacc[j] = wmma_bf16(qa0, kb0, sacc[j]);
      sacc[j] = wmma_bf16(qa1, kb1, sacc[j]);
    }

    // online softmax; row stats via DPP16 half-wave reductions
    float mnew[8], alpha[8], rs[8];
#pragma unroll
    for (int r = 0; r < 8; ++r) {
      float mx = fmaxf(fmaxf(sacc[0][r], sacc[1][r]),
                       fmaxf(sacc[2][r], sacc[3][r]));
      mx = red_max16(mx);
      mnew[r]  = fmaxf(m_i[r], mx);
      alpha[r] = __expf(m_i[r] - mnew[r]);
    }
#pragma unroll
    for (int r = 0; r < 8; ++r) rs[r] = 0.0f;
#pragma unroll
    for (int j = 0; j < 4; ++j)
#pragma unroll
      for (int r = 0; r < 8; ++r) {
        float p = __expf(sacc[j][r] - mnew[r]);
        sacc[j][r] = p;
        rs[r] += p;
      }
#pragma unroll
    for (int r = 0; r < 8; ++r) {
      rs[r]  = red_sum16(rs[r]);
      l_i[r] = l_i[r] * alpha[r] + rs[r];
      m_i[r] = mnew[r];
    }
#pragma unroll
    for (int j = 0; j < 4; ++j)
#pragma unroll
      for (int r = 0; r < 8; ++r) acc_o[j][r] *= alpha[r];

    // C-layout -> A-layout for P via wave-private LDS (in-order per wave)
#pragma unroll
    for (int j = 0; j < 4; ++j)
#pragma unroll
      for (int r = 0; r < 8; ++r)
        pw[(r + mh) * 64 + j * 16 + nl] = (bf16)sacc[j][r];

    v16bf pa0 = load_frag(pw,      64);
    v16bf pa1 = load_frag(pw + 32, 64);
#pragma unroll
    for (int j = 0; j < 4; ++j) {
      v16bf vb0 = load_frag(&Vt[cur][(j * 16) * 64],      64);
      v16bf vb1 = load_frag(&Vt[cur][(j * 16) * 64 + 32], 64);
      acc_o[j] = wmma_bf16(pa0, vb0, acc_o[j]);
      acc_o[j] = wmma_bf16(pa1, vb1, acc_o[j]);
    }
  }

  // normalize, write ctx row-major [B*S, D] bf16 (col = h*HD + d)
  float inv[8];
#pragma unroll
  for (int r = 0; r < 8; ++r) inv[r] = 1.0f / l_i[r];
  const size_t row0 = (size_t)b * S_ + blockIdx.x * 128 + wave * 16;
#pragma unroll
  for (int j = 0; j < 4; ++j)
#pragma unroll
    for (int r = 0; r < 8; ++r) {
      float o = acc_o[j][r] * inv[r];
      ctx[(row0 + r + mh) * D_ + (size_t)h * HD_ + j * 16 + nl] = (bf16)o;
    }
}

// ---------------------------------------------------------------------------
extern "C" void kernel_launch(void* const* d_in, const int* in_sizes, int n_in,
                              void* d_out, int out_size, void* d_ws, size_t ws_size,
                              hipStream_t stream) {
  const float* in_q = (const float*)d_in[0];
  const float* in_k = (const float*)d_in[1];
  const float* in_v = (const float*)d_in[2];
  const float* Wq   = (const float*)d_in[3];
  const float* bq   = (const float*)d_in[4];
  const float* Wk   = (const float*)d_in[5];
  const float* bk   = (const float*)d_in[6];
  const float* Wv   = (const float*)d_in[7];
  const float* bv   = (const float*)d_in[8];
  const float* Wo   = (const float*)d_in[9];
  const float* bo   = (const float*)d_in[10];
  float* out = (float*)d_out;

  const size_t XN = (size_t)M_ * D_;
  const size_t WN = (size_t)D_ * D_;
  bf16* p  = (bf16*)d_ws;
  bf16* Xq = p; p += XN;
  bf16* Xk = p; p += XN;
  bf16* Xv = p; p += XN;
  bf16* wq = p; p += WN;
  bf16* wk = p; p += WN;
  bf16* wv = p; p += WN;
  bf16* wo = p; p += WN;
  bf16* qh = p; p += XN;               // [B,H,S,HD]
  bf16* kh = p; p += XN;               // [B,H,S,HD]
  bf16* vT = p; p += XN;               // [B,H,HD,S]
  bf16* cx = p; p += XN;               // ctx [B*S, D]

  auto cvt = [&](const float* s, bf16* d, size_t n) {
    int n4 = (int)(n / 4);
    cvt_f32_bf16<<<(n4 + 255) / 256, 256, 0, stream>>>(
        (const float4*)s, (v4bf*)d, n4);
  };
  cvt(in_q, Xq, XN); cvt(in_k, Xk, XN); cvt(in_v, Xv, XN);
  cvt(Wq, wq, WN);   cvt(Wk, wk, WN);  cvt(Wv, wv, WN); cvt(Wo, wo, WN);

  dim3 gg(D_ / 64, M_ / 128);          // 16 x 32 workgroups
  gemm_bf16_kernel<0><<<gg, 256, 0, stream>>>(Xq, wq, bq, qh, (float*)nullptr);
  gemm_bf16_kernel<1><<<gg, 256, 0, stream>>>(Xk, wk, bk, kh, (float*)nullptr);
  gemm_bf16_kernel<2><<<gg, 256, 0, stream>>>(Xv, wv, bv, vT, (float*)nullptr);

  attn_kernel<<<dim3(S_ / 128, B_ * H_), 256, 0, stream>>>(qh, kh, vT, cx);

  gemm_bf16_kernel<3><<<gg, 256, 0, stream>>>(cx, wo, bo, (bf16*)nullptr, out);
}
